// CMPNNLayer_80685255622666
// MI455X (gfx1250) — matrix operand order repaired and checked by reading, not compiled, not run on hardware
//
#include <hip/hip_runtime.h>

// ---------------------------------------------------------------------------
// CMPNN layer, fully fused for gfx1250 (wave32, WMMA f32_16x16x32_f16).
// ---------------------------------------------------------------------------

typedef __attribute__((ext_vector_type(16))) _Float16 v16h;
typedef __attribute__((ext_vector_type(8)))  _Float16 v8h;
typedef __attribute__((ext_vector_type(8)))  float    v8f;

#define NDIM 64
#define EDIM 64
#define HDIM 192
#define NNODE 50000
#define NEDGE 800000
#define TM 64          // rows per block tile
#define AGGW 128       // ND + ED
#define AGG2 256       // 2*(ND+ED)
#define LN_EPS 1e-5f

#if defined(__has_builtin) && __has_builtin(__builtin_amdgcn_sched_barrier)
#define SCHED_FENCE() __builtin_amdgcn_sched_barrier(0)
#else
#define SCHED_FENCE()
#endif

// ---------------------------------------------------------------------------
// WMMA fragment helpers
// ---------------------------------------------------------------------------

// A-matrix 16x32 f16 fragment (per CDNA5 layout):
//   lane<16 : row M=lane,    K = {k0+0..7, k0+16..23}
//   lane>=16: row M=lane-16, K = {k0+8..15, k0+24..31}
__device__ __forceinline__ v16h load_a(const _Float16* X, int stride,
                                       int mbase, int k0, int lane) {
  int row = mbase + (lane & 15);
  int h8  = (lane >> 4) << 3;              // 0 or 8 halves
  const _Float16* p = X + row * stride + k0 + h8;
  v8h lo = *(const v8h*)p;                 // K = k0+h8 .. +7
  v8h hi = *(const v8h*)(p + 16);          // K = k0+16+h8 .. +7
  v16h a;
#pragma unroll
  for (int i = 0; i < 8; ++i) { a[i] = lo[i]; a[i + 8] = hi[i]; }
  return a;
}

// B-matrix 32x16 f16 fragment, B[k][n] = W[n][k] with W row-major [Nout][Kin]:
//   lane<16 : col N=lane,    K = {k0+0..7, k0+16..23}
//   lane>=16: col N=lane-16, K = {k0+8..15, k0+24..31}
__device__ __forceinline__ v16h load_b(const _Float16* W, int stride,
                                       int nbase, int k0, int lane) {
  int col = nbase + (lane & 15);
  int kk  = k0 + ((lane >> 4) << 3);
  const _Float16* p = W + col * stride + kk;
  v8h lo = *(const v8h*)p;
  v8h hi = *(const v8h*)(p + 16);
  v16h b;
#pragma unroll
  for (int i = 0; i < 8; ++i) { b[i] = lo[i]; b[i + 8] = hi[i]; }
  return b;
}

__device__ __forceinline__ v8f wmma16(v16h a, v16h b, v8f c) {
  return __builtin_amdgcn_wmma_f32_16x16x32_f16(
      /*neg_a=*/false, a, /*neg_b=*/false, b,
      /*c_mod=*/(short)0, c, /*reuse_a=*/false, /*reuse_b=*/false);
}

__device__ __forceinline__ v8f vzero8() {
  v8f z = {0.f, 0.f, 0.f, 0.f, 0.f, 0.f, 0.f, 0.f};
  return z;
}

__device__ __forceinline__ float sigm(float x) {
  return 1.0f / (1.0f + __expf(-x));
}

__device__ __forceinline__ float lrelu(float x, float s) {
  return x > 0.0f ? x : s * x;
}

// Order-consistent float atomic max (buffer initialized to -inf bit pattern).
__device__ __forceinline__ void atomicMaxF(float* addr, float v) {
  if (v >= 0.0f) atomicMax((int*)addr, __float_as_int(v));
  else           atomicMin((unsigned int*)addr, __float_as_uint(v));
}

// ---------------------------------------------------------------------------
// Prep kernels
// ---------------------------------------------------------------------------

__global__ void f32_to_f16_kernel(const float* __restrict__ src,
                                  _Float16* __restrict__ dst, int n) {
  for (int i = blockIdx.x * blockDim.x + threadIdx.x; i < n;
       i += gridDim.x * blockDim.x)
    dst[i] = (_Float16)src[i];
}

__global__ void init_agg_kernel(float* __restrict__ agg_sum,
                                unsigned int* __restrict__ agg_max_bits) {
  int i = blockIdx.x * blockDim.x + threadIdx.x;
  if (i < NNODE * AGGW) {
    agg_sum[i] = 0.0f;
    agg_max_bits[i] = 0xFF800000u;   // -inf
  }
}

// ---------------------------------------------------------------------------
// Edge kernel: booster + GRU + residual + scatter aggregation, 64 edges/block
// ---------------------------------------------------------------------------

__global__ __launch_bounds__(128) void edge_kernel(
    const float* __restrict__ node_feats, const float* __restrict__ edge_feats,
    const int* __restrict__ src, const int* __restrict__ dst,
    const _Float16* __restrict__ W1, const float* __restrict__ b1,
    const float* __restrict__ ln_g, const float* __restrict__ ln_b,
    const _Float16* __restrict__ W2, const float* __restrict__ b2,
    const _Float16* __restrict__ Wih, const _Float16* __restrict__ Whh,
    const float* __restrict__ bih, const float* __restrict__ bhh,
    const _Float16* __restrict__ Wre, const float* __restrict__ bre,
    float* __restrict__ out_edges,
    float* __restrict__ agg_sum, float* __restrict__ agg_max) {
  __shared__ _Float16 Xa[TM * HDIM];   // msg, later GEMM2 output (f16)
  __shared__ _Float16 Xb[TM * HDIM];   // LN output (f16)
  __shared__ _Float16 Hs[TM * EDIM];   // edge feats (f16)
  __shared__ float    Yf[TM * HDIM];   // GEMM1 output (f32) for LayerNorm
  __shared__ float    Rg[TM * EDIM];   // r gate
  __shared__ float    Zg[TM * EDIM];   // z gate
  __shared__ float    LnS[TM * 2];     // LN partial sums
  __shared__ float    LnS2[TM * 2];

  const int tid  = threadIdx.x;
  const int lane = tid & 31;
  const int wave = tid >> 5;
  const int e0   = blockIdx.x * TM;
  const int mb   = wave * 16;

  // ---- gather msg = [h_src, h_dst, e]; also scatter h_src into aggregation
  for (int i = tid; i < TM * HDIM; i += 128) {
    int r = i / HDIM, c = i % HDIM;
    int e = e0 + r;
    float v;
    if (c < NDIM) {
      v = node_feats[src[e] * NDIM + c];
      int d = dst[e];
      atomicAdd(&agg_sum[d * AGGW + c], v);
      atomicMaxF(&agg_max[d * AGGW + c], v);
    } else if (c < 2 * NDIM) {
      v = node_feats[dst[e] * NDIM + (c - NDIM)];
    } else {
      v = edge_feats[e * EDIM + (c - 2 * NDIM)];
      Hs[r * EDIM + (c - 2 * NDIM)] = (_Float16)v;
    }
    Xa[i] = (_Float16)v;
  }
  __syncthreads();

  // ---- GEMM1: Yf = msg @ W1^T + b1
  {
    v16h a[6];
#pragma unroll
    for (int kt = 0; kt < 6; ++kt) a[kt] = load_a(Xa, HDIM, mb, kt * 32, lane);
#pragma unroll
    for (int nt = 0; nt < HDIM / 16; ++nt) {
      v16h b[6];
#pragma unroll
      for (int kt = 0; kt < 6; ++kt)
        b[kt] = load_b(W1, HDIM, nt * 16, kt * 32, lane);
      SCHED_FENCE();
      v8f acc = vzero8();
#pragma unroll
      for (int kt = 0; kt < 6; ++kt) acc = wmma16(a[kt], b[kt], acc);
      int n = nt * 16 + (lane & 15);
      int mrow = mb + ((lane >> 4) << 3);
      float bb = b1[n];
#pragma unroll
      for (int r = 0; r < 8; ++r) Yf[(mrow + r) * HDIM + n] = acc[r] + bb;
    }
  }
  __syncthreads();

  // ---- LayerNorm + LeakyReLU(0.2) -> Xb (f16); 2 threads per row
  {
    int row = tid >> 1, half = tid & 1;
    int c0 = half * (HDIM / 2);
    float s = 0.f, s2 = 0.f;
    for (int c = c0; c < c0 + HDIM / 2; ++c) {
      float v = Yf[row * HDIM + c];
      s += v; s2 += v * v;
    }
    LnS[tid] = s;
    LnS2[tid] = s2;
    __syncthreads();
    float ts  = LnS[row * 2] + LnS[row * 2 + 1];
    float ts2 = LnS2[row * 2] + LnS2[row * 2 + 1];
    float mu = ts / HDIM;
    float rs = rsqrtf(ts2 / HDIM - mu * mu + LN_EPS);
    for (int c = c0; c < c0 + HDIM / 2; ++c) {
      float v = (Yf[row * HDIM + c] - mu) * rs * ln_g[c] + ln_b[c];
      Xb[row * HDIM + c] = (_Float16)lrelu(v, 0.2f);
    }
  }
  __syncthreads();

  // ---- GEMM2: Xa = f16(Xb @ W2^T + b2)
  {
    v16h a[6];
#pragma unroll
    for (int kt = 0; kt < 6; ++kt) a[kt] = load_a(Xb, HDIM, mb, kt * 32, lane);
#pragma unroll
    for (int nt = 0; nt < HDIM / 16; ++nt) {
      v16h b[6];
#pragma unroll
      for (int kt = 0; kt < 6; ++kt)
        b[kt] = load_b(W2, HDIM, nt * 16, kt * 32, lane);
      SCHED_FENCE();
      v8f acc = vzero8();
#pragma unroll
      for (int kt = 0; kt < 6; ++kt) acc = wmma16(a[kt], b[kt], acc);
      int n = nt * 16 + (lane & 15);
      int mrow = mb + ((lane >> 4) << 3);
      float bb = b2[n];
#pragma unroll
      for (int r = 0; r < 8; ++r)
        Xa[(mrow + r) * HDIM + n] = (_Float16)(acc[r] + bb);
    }
  }
  __syncthreads();

  // ---- GRU (gi from Xa K=192, gh from Hs K=64) + Wre residual + scatter
  {
    v16h a[6], ah[2];
#pragma unroll
    for (int kt = 0; kt < 6; ++kt) a[kt] = load_a(Xa, HDIM, mb, kt * 32, lane);
#pragma unroll
    for (int kt = 0; kt < 2; ++kt) ah[kt] = load_a(Hs, EDIM, mb, kt * 32, lane);

#pragma unroll
    for (int nt = 0; nt < HDIM / 16; ++nt) {
      v16h b[6], bh[2];
#pragma unroll
      for (int kt = 0; kt < 6; ++kt)
        b[kt] = load_b(Wih, HDIM, nt * 16, kt * 32, lane);
#pragma unroll
      for (int kt = 0; kt < 2; ++kt)
        bh[kt] = load_b(Whh, EDIM, nt * 16, kt * 32, lane);
      v16h br_[2];
      if (nt >= 8) {
#pragma unroll
        for (int kt = 0; kt < 2; ++kt)
          br_[kt] = load_b(Wre, EDIM, (nt - 8) * 16, kt * 32, lane);
      }
      SCHED_FENCE();
      v8f gi = vzero8(), gh = vzero8();
#pragma unroll
      for (int kt = 0; kt < 6; ++kt) gi = wmma16(a[kt], b[kt], gi);
#pragma unroll
      for (int kt = 0; kt < 2; ++kt) gh = wmma16(ah[kt], bh[kt], gh);

      int n = nt * 16 + (lane & 15);
      int mrow = mb + ((lane >> 4) << 3);
      float bi = bih[n], bhv = bhh[n];
      if (nt < 4) {                       // r gate, cols 0..63
#pragma unroll
        for (int r = 0; r < 8; ++r)
          Rg[(mrow + r) * EDIM + n] = sigm(gi[r] + bi + gh[r] + bhv);
      } else if (nt < 8) {                // z gate, cols 64..127
#pragma unroll
        for (int r = 0; r < 8; ++r)
          Zg[(mrow + r) * EDIM + (n - EDIM)] = sigm(gi[r] + bi + gh[r] + bhv);
      } else {                            // n gate + output, cols 128..191
        int j = n - 2 * EDIM;
        v8f res = vzero8();
#pragma unroll
        for (int kt = 0; kt < 2; ++kt) res = wmma16(ah[kt], br_[kt], res);
        float br = bre[j];
#pragma unroll
        for (int r = 0; r < 8; ++r) {
          int m = mrow + r;
          float rv = Rg[m * EDIM + j];
          float zv = Zg[m * EDIM + j];
          float hn = tanhf(gi[r] + bi + rv * (gh[r] + bhv));
          float hp = (float)Hs[m * EDIM + j];
          float hv = (1.0f - zv) * hn + zv * hp;
          float o  = lrelu(hv + res[r] + br, 0.01f);
          int e = e0 + m;
          out_edges[e * EDIM + j] = o;
          int d = dst[e];
          atomicAdd(&agg_sum[d * AGGW + NDIM + j], o);
          atomicMaxF(&agg_max[d * AGGW + NDIM + j], o);
        }
      }
    }
  }
}

// ---------------------------------------------------------------------------
// Node kernel: booster(K=256) + GRU + Wra residual, 64 nodes/block
// ---------------------------------------------------------------------------

__global__ __launch_bounds__(128) void node_kernel(
    const float* __restrict__ node_feats,
    const float* __restrict__ agg_sum, const float* __restrict__ agg_max,
    const _Float16* __restrict__ W1, const float* __restrict__ b1,
    const float* __restrict__ ln_g, const float* __restrict__ ln_b,
    const _Float16* __restrict__ W2, const float* __restrict__ b2,
    const _Float16* __restrict__ Wih, const _Float16* __restrict__ Whh,
    const float* __restrict__ bih, const float* __restrict__ bhh,
    const _Float16* __restrict__ Wra, const float* __restrict__ bra,
    float* __restrict__ out_nodes) {
  __shared__ _Float16 Xa[TM * AGG2];   // agg input (K=256); later GEMM2 out
  __shared__ _Float16 Xb[TM * HDIM];
  __shared__ _Float16 Hs[TM * NDIM];
  __shared__ float    Yf[TM * HDIM];
  __shared__ float    Rg[TM * NDIM];
  __shared__ float    Zg[TM * NDIM];
  __shared__ float    LnS[TM * 2];
  __shared__ float    LnS2[TM * 2];

  const int tid  = threadIdx.x;
  const int lane = tid & 31;
  const int wave = tid >> 5;
  const int n0   = blockIdx.x * TM;
  const int mb   = wave * 16;

  // ---- build agg = [sum, max(fixed)] (f16) and h (f16)
  for (int i = tid; i < TM * AGG2; i += 128) {
    int r = i / AGG2, c = i % AGG2;
    int node = n0 + r;
    float v = 0.0f;
    if (node < NNODE) {
      if (c < AGGW) {
        v = agg_sum[node * AGGW + c];
      } else {
        v = agg_max[node * AGGW + (c - AGGW)];
        if (!(v >= -3.0e38f)) v = 0.0f;   // -inf (empty mailbox) or NaN -> 0
      }
    }
    Xa[i] = (_Float16)v;
  }
  for (int i = tid; i < TM * NDIM; i += 128) {
    int r = i / NDIM, c = i % NDIM;
    int node = n0 + r;
    Hs[i] = (_Float16)(node < NNODE ? node_feats[node * NDIM + c] : 0.0f);
  }
  __syncthreads();

  // ---- GEMM1: Yf = agg @ W1n^T + b1   (K=256)
  {
    v16h a[8];
#pragma unroll
    for (int kt = 0; kt < 8; ++kt) a[kt] = load_a(Xa, AGG2, mb, kt * 32, lane);
#pragma unroll
    for (int nt = 0; nt < HDIM / 16; ++nt) {
      v16h b[8];
#pragma unroll
      for (int kt = 0; kt < 8; ++kt)
        b[kt] = load_b(W1, AGG2, nt * 16, kt * 32, lane);
      SCHED_FENCE();
      v8f acc = vzero8();
#pragma unroll
      for (int kt = 0; kt < 8; ++kt) acc = wmma16(a[kt], b[kt], acc);
      int n = nt * 16 + (lane & 15);
      int mrow = mb + ((lane >> 4) << 3);
      float bb = b1[n];
#pragma unroll
      for (int r = 0; r < 8; ++r) Yf[(mrow + r) * HDIM + n] = acc[r] + bb;
    }
  }
  __syncthreads();

  // ---- LayerNorm + LeakyReLU(0.2) -> Xb; 2 threads per row
  {
    int row = tid >> 1, half = tid & 1;
    int c0 = half * (HDIM / 2);
    float s = 0.f, s2 = 0.f;
    for (int c = c0; c < c0 + HDIM / 2; ++c) {
      float v = Yf[row * HDIM + c];
      s += v; s2 += v * v;
    }
    LnS[tid] = s;
    LnS2[tid] = s2;
    __syncthreads();
    float ts  = LnS[row * 2] + LnS[row * 2 + 1];
    float ts2 = LnS2[row * 2] + LnS2[row * 2 + 1];
    float mu = ts / HDIM;
    float rs = rsqrtf(ts2 / HDIM - mu * mu + LN_EPS);
    for (int c = c0; c < c0 + HDIM / 2; ++c) {
      float v = (Yf[row * HDIM + c] - mu) * rs * ln_g[c] + ln_b[c];
      Xb[row * HDIM + c] = (_Float16)lrelu(v, 0.2f);
    }
  }
  __syncthreads();

  // ---- GEMM2: Xa(reused, stride HDIM) = f16(Xb @ W2^T + b2)
  {
    v16h a[6];
#pragma unroll
    for (int kt = 0; kt < 6; ++kt) a[kt] = load_a(Xb, HDIM, mb, kt * 32, lane);
#pragma unroll
    for (int nt = 0; nt < HDIM / 16; ++nt) {
      v16h b[6];
#pragma unroll
      for (int kt = 0; kt < 6; ++kt)
        b[kt] = load_b(W2, HDIM, nt * 16, kt * 32, lane);
      SCHED_FENCE();
      v8f acc = vzero8();
#pragma unroll
      for (int kt = 0; kt < 6; ++kt) acc = wmma16(a[kt], b[kt], acc);
      int n = nt * 16 + (lane & 15);
      int mrow = mb + ((lane >> 4) << 3);
      float bb = b2[n];
#pragma unroll
      for (int r = 0; r < 8; ++r)
        Xa[(mrow + r) * HDIM + n] = (_Float16)(acc[r] + bb);
    }
  }
  __syncthreads();

  // ---- GRU + Wra residual -> update_nodes
  {
    v16h a[6], ah[2];
#pragma unroll
    for (int kt = 0; kt < 6; ++kt) a[kt] = load_a(Xa, HDIM, mb, kt * 32, lane);
#pragma unroll
    for (int kt = 0; kt < 2; ++kt) ah[kt] = load_a(Hs, NDIM, mb, kt * 32, lane);

#pragma unroll
    for (int nt = 0; nt < HDIM / 16; ++nt) {
      v16h b[6], bh[2];
#pragma unroll
      for (int kt = 0; kt < 6; ++kt)
        b[kt] = load_b(Wih, HDIM, nt * 16, kt * 32, lane);
#pragma unroll
      for (int kt = 0; kt < 2; ++kt)
        bh[kt] = load_b(Whh, NDIM, nt * 16, kt * 32, lane);
      v16h br_[2];
      if (nt >= 8) {
#pragma unroll
        for (int kt = 0; kt < 2; ++kt)
          br_[kt] = load_b(Wra, NDIM, (nt - 8) * 16, kt * 32, lane);
      }
      SCHED_FENCE();
      v8f gi = vzero8(), gh = vzero8();
#pragma unroll
      for (int kt = 0; kt < 6; ++kt) gi = wmma16(a[kt], b[kt], gi);
#pragma unroll
      for (int kt = 0; kt < 2; ++kt) gh = wmma16(ah[kt], bh[kt], gh);

      int n = nt * 16 + (lane & 15);
      int mrow = mb + ((lane >> 4) << 3);
      float bi = bih[n], bhv = bhh[n];
      if (nt < 4) {
#pragma unroll
        for (int r = 0; r < 8; ++r)
          Rg[(mrow + r) * NDIM + n] = sigm(gi[r] + bi + gh[r] + bhv);
      } else if (nt < 8) {
#pragma unroll
        for (int r = 0; r < 8; ++r)
          Zg[(mrow + r) * NDIM + (n - NDIM)] = sigm(gi[r] + bi + gh[r] + bhv);
      } else {
        int j = n - 2 * NDIM;
        v8f res = vzero8();
#pragma unroll
        for (int kt = 0; kt < 2; ++kt) res = wmma16(ah[kt], br_[kt], res);
        float br = bra[j];
#pragma unroll
        for (int r = 0; r < 8; ++r) {
          int m = mrow + r;
          int node = n0 + m;
          if (node < NNODE) {
            float rv = Rg[m * NDIM + j];
            float zv = Zg[m * NDIM + j];
            float hn = tanhf(gi[r] + bi + rv * (gh[r] + bhv));
            float hp = (float)Hs[m * NDIM + j];
            float hv = (1.0f - zv) * hn + zv * hp;
            out_nodes[node * NDIM + j] = lrelu(hv + res[r] + br, 0.01f);
          }
        }
      }
    }
  }
}

// ---------------------------------------------------------------------------
// Host launcher
// ---------------------------------------------------------------------------

extern "C" void kernel_launch(void* const* d_in, const int* in_sizes, int n_in,
                              void* d_out, int out_size, void* d_ws,
                              size_t ws_size, hipStream_t stream) {
  (void)in_sizes; (void)n_in; (void)out_size; (void)ws_size;

  const float* node_feats = (const float*)d_in[0];
  const float* edge_feats = (const float*)d_in[1];
  const int*   src        = (const int*)d_in[2];
  const int*   dst        = (const int*)d_in[3];
  const float* W1e  = (const float*)d_in[4];
  const float* W1n  = (const float*)d_in[5];
  const float* b1   = (const float*)d_in[6];
  const float* ln_g = (const float*)d_in[7];
  const float* ln_b = (const float*)d_in[8];
  const float* W2   = (const float*)d_in[9];
  const float* b2   = (const float*)d_in[10];
  const float* Wih_e = (const float*)d_in[11];
  const float* Whh_e = (const float*)d_in[12];
  const float* bih_e = (const float*)d_in[13];
  const float* bhh_e = (const float*)d_in[14];
  const float* Wih_a = (const float*)d_in[15];
  const float* Whh_a = (const float*)d_in[16];
  const float* bih_a = (const float*)d_in[17];
  const float* bhh_a = (const float*)d_in[18];
  const float* Wre = (const float*)d_in[19];
  const float* bre = (const float*)d_in[20];
  const float* Wra = (const float*)d_in[21];
  const float* bra = (const float*)d_in[22];

  float* out_nodes = (float*)d_out;                        // [N, ND]
  float* out_edges = (float*)d_out + (size_t)NNODE * NDIM; // [E, ED]

  // --- workspace carve-up (256B aligned chunks)
  char* ws = (char*)d_ws;
  size_t off = 0;
  auto take = [&](size_t bytes) -> void* {
    void* p = ws + off;
    off += (bytes + 255) & ~(size_t)255;
    return p;
  };
  float* agg_sum = (float*)take((size_t)NNODE * AGGW * sizeof(float));
  float* agg_max = (float*)take((size_t)NNODE * AGGW * sizeof(float));
  _Float16* W1e16   = (_Float16*)take((size_t)HDIM * HDIM * 2);
  _Float16* W216    = (_Float16*)take((size_t)HDIM * HDIM * 2);
  _Float16* Wih_e16 = (_Float16*)take((size_t)HDIM * HDIM * 2);
  _Float16* Whh_e16 = (_Float16*)take((size_t)HDIM * EDIM * 2);
  _Float16* Wre16   = (_Float16*)take((size_t)EDIM * EDIM * 2);
  _Float16* W1n16   = (_Float16*)take((size_t)HDIM * AGG2 * 2);
  _Float16* Wih_a16 = (_Float16*)take((size_t)HDIM * HDIM * 2);
  _Float16* Whh_a16 = (_Float16*)take((size_t)HDIM * NDIM * 2);
  _Float16* Wra16   = (_Float16*)take((size_t)NDIM * NDIM * 2);

  // --- weight conversion (tiny; L2-resident afterwards)
  auto conv = [&](const float* s, _Float16* d, int n) {
    f32_to_f16_kernel<<<64, 256, 0, stream>>>(s, d, n);
  };
  conv(W1e,   W1e16,   HDIM * HDIM);
  conv(W2,    W216,    HDIM * HDIM);
  conv(Wih_e, Wih_e16, HDIM * HDIM);
  conv(Whh_e, Whh_e16, HDIM * EDIM);
  conv(Wre,   Wre16,   EDIM * EDIM);
  conv(W1n,   W1n16,   HDIM * AGG2);
  conv(Wih_a, Wih_a16, HDIM * HDIM);
  conv(Whh_a, Whh_a16, HDIM * NDIM);
  conv(Wra,   Wra16,   NDIM * NDIM);

  // --- init aggregation buffers
  init_agg_kernel<<<(NNODE * AGGW + 255) / 256, 256, 0, stream>>>(
      agg_sum, (unsigned int*)agg_max);

  // --- edge pipeline (also scatters into agg buffers)
  edge_kernel<<<(NEDGE + TM - 1) / TM, 128, 0, stream>>>(
      node_feats, edge_feats, src, dst,
      W1e16, b1, ln_g, ln_b, W216, b2,
      Wih_e16, Whh_e16, bih_e, bhh_e,
      Wre16, bre,
      out_edges, agg_sum, agg_max);

  // --- node pipeline
  node_kernel<<<(NNODE + TM - 1) / TM, 128, 0, stream>>>(
      node_feats, agg_sum, agg_max,
      W1n16, b1, ln_g, ln_b, W216, b2,
      Wih_a16, Whh_a16, bih_a, bhh_a,
      Wra16, bra,
      out_nodes);
}
